// SgCLLoss_34299608826247
// MI455X (gfx1250) — matrix operand order repaired and checked by reading, not compiled, not run on hardware
//
#include <hip/hip_runtime.h>
#include <math.h>

// ---------------------------------------------------------------------------
// SgCL loss, fused:  loss = mean_{v,n}[ log(sum_m (1-S[n,m]) e^{sim[v,n,m]/T})
//                                       - sim[v,n,n]/T ]
// sim = normalize(Hc) @ normalize(Hv[v])^T via v_wmma_f32_16x16x32_f16,
// reduced in the epilogue (sim never stored).  S (256 MB) is streamed exactly
// once with non-temporal loads; Hv tiles are double-buffered in LDS via the
// CDNA5 async global->LDS path (global_load_async_to_lds_b128 / ASYNCcnt).
// ---------------------------------------------------------------------------

typedef __attribute__((ext_vector_type(16))) _Float16 v16h;
typedef __attribute__((ext_vector_type(8)))  _Float16 v8h;
typedef __attribute__((ext_vector_type(4)))  _Float16 v4h;
typedef __attribute__((ext_vector_type(8)))  float    v8f;
typedef __attribute__((ext_vector_type(4)))  float    v4f;
typedef __attribute__((ext_vector_type(4)))  int      v4i;

#define NROWS  8192
#define DIM    128
#define NVIEWS 3
#define MSPLIT 8          // m-range split for grid parallelism
#define INV_T  2.0f       // 1 / TEMPERATURE
#define EPS_D  1e-9f
#define PADW   136        // padded LDS row stride (halves): bank-conflict-free
#define TILE_HALVES (NVIEWS * 16 * PADW)

union V16 { v16h v; v8h h[2]; };

#if __has_builtin(__builtin_amdgcn_global_load_async_to_lds_b128)
#define HAS_ASYNC_LDS 1
typedef __attribute__((address_space(1))) v4i as1_v4i;   // global 16B vector
typedef __attribute__((address_space(3))) v4i as3_v4i;   // LDS    16B vector
#else
#define HAS_ASYNC_LDS 0
#endif

__device__ __forceinline__ void wait_async_copies() {
#if HAS_ASYNC_LDS
#if __has_builtin(__builtin_amdgcn_s_wait_asynccnt)
    __builtin_amdgcn_s_wait_asynccnt(0);
#else
    asm volatile("s_wait_asynccnt 0x0" ::: "memory");
#endif
#endif
}

__device__ __forceinline__ float half16_sum(float x) {
    // sum across a 16-lane group (masks < 16 stay within the half-wave)
    x += __shfl_xor(x, 1, 32);
    x += __shfl_xor(x, 2, 32);
    x += __shfl_xor(x, 4, 32);
    x += __shfl_xor(x, 8, 32);
    return x;
}
__device__ __forceinline__ float wave32_sum(float x) {
    x = half16_sum(x);
    x += __shfl_xor(x, 16, 32);
    return x;
}

// Cooperative fill of one 16-wide m-tile of Hv16 (all 3 views) into LDS.
// 256 threads x 3 x 16B = 12 KB.  Rows padded to PADW halves.
__device__ __forceinline__ void copy_tile(
        _Float16* __restrict__ dst, const _Float16* __restrict__ Hv16,
        int mBase, int tid) {
    const int row   = tid >> 4;    // 0..15
    const int chunk = tid & 15;    // 0..15 (16B chunks of a 256B row)
    #pragma unroll
    for (int v = 0; v < NVIEWS; ++v) {
        const _Float16* g = Hv16 + ((size_t)v * NROWS + mBase + row) * DIM + chunk * 8;
        _Float16*       l = dst + (v * 16 + row) * PADW + chunk * 8;
#if HAS_ASYNC_LDS
        __builtin_amdgcn_global_load_async_to_lds_b128(
            (as1_v4i*)(void*)(const void*)g, (as3_v4i*)(void*)l, 0, 0);
#else
        *(v8h*)l = *(const v8h*)g;
#endif
    }
}

// --- Kernel 1: L2-normalize rows of Hc and Hv, emit f16 copies in ws -------
__global__ void sgcl_normalize_f16(const float* __restrict__ Hc,
                                   const float* __restrict__ Hv,
                                   _Float16* __restrict__ Hc16,
                                   _Float16* __restrict__ Hv16) {
    const int wave = threadIdx.x >> 5;
    const int lane = threadIdx.x & 31;
    const int row  = blockIdx.x * 8 + wave;            // 0 .. 4*NROWS-1

    const float* src;
    _Float16*    dst;
    if (row < NROWS) {
        src = Hc   + (size_t)row * DIM;
        dst = Hc16 + (size_t)row * DIM;
    } else {
        const size_t r = (size_t)(row - NROWS);
        src = Hv   + r * DIM;
        dst = Hv16 + r * DIM;
    }

    v4f x = *(const v4f*)(src + lane * 4);
    float ss = x.x * x.x + x.y * x.y + x.z * x.z + x.w * x.w;
    ss = wave32_sum(ss);
    const float scale = 1.0f / fmaxf(sqrtf(ss), 1e-12f);

    v4h h;
    h.x = (_Float16)(x.x * scale);
    h.y = (_Float16)(x.y * scale);
    h.z = (_Float16)(x.z * scale);
    h.w = (_Float16)(x.w * scale);
    *(v4h*)(dst + lane * 4) = h;
}

// --- Kernel 2: positives pos[v*N+n] = <HcN[n], HvN[v,n]> (f32, exact) ------
__global__ void sgcl_positives(const float* __restrict__ Hc,
                               const float* __restrict__ Hv,
                               float* __restrict__ pos) {
    const int wave = threadIdx.x >> 5;
    const int lane = threadIdx.x & 31;
    const int idx  = blockIdx.x * 8 + wave;            // v*NROWS + n
    const int n    = idx & (NROWS - 1);

    v4f a = *(const v4f*)(Hc + (size_t)n   * DIM + lane * 4);
    v4f b = *(const v4f*)(Hv + (size_t)idx * DIM + lane * 4);

    float na = a.x * a.x + a.y * a.y + a.z * a.z + a.w * a.w;
    float nb = b.x * b.x + b.y * b.y + b.z * b.z + b.w * b.w;
    float d  = a.x * b.x + a.y * b.y + a.z * b.z + a.w * b.w;
    na = wave32_sum(na);
    nb = wave32_sum(nb);
    d  = wave32_sum(d);
    if (lane == 0)
        pos[idx] = d / (fmaxf(sqrtf(na), 1e-12f) * fmaxf(sqrtf(nb), 1e-12f));
}

// --- Kernel 3: fused WMMA GEMM + weighted-softmax denominator --------------
// grid (64, MSPLIT), 256 thr (8 waves).  Each wave owns 16 rows n; each
// blockIdx.y slice covers NROWS/MSPLIT columns m.  Hv tiles double-buffered
// in LDS (async fill for tile t+1 overlaps WMMAs of tile t).
__global__ void __launch_bounds__(256, 1)
sgcl_denominator(const float* __restrict__ S,
                 const _Float16* __restrict__ Hc16,
                 const _Float16* __restrict__ Hv16,
                 float* __restrict__ denomP) {
    __shared__ __align__(16) _Float16 smem[2][TILE_HALVES];

    const int tid   = threadIdx.x;
    const int wave  = tid >> 5;
    const int lane  = tid & 31;
    const int col   = lane & 15;       // A-row / B-col / C-col index
    const int hi    = lane >> 4;       // half-wave selector
    const int nBase = blockIdx.x * 128 + wave * 16;
    const int mStart  = blockIdx.y * (NROWS / MSPLIT);
    const int nTiles  = (NROWS / MSPLIT) / 16;

    // A fragments (16x32 f16): lane holds row (nBase+col); per K-chunk kc the
    // lane's halves are [kc + hi*8, +8) and [kc+16 + hi*8, +8)  (ISA 7.12.2).
    V16 afrag[4];
    {
        const _Float16* arow = Hc16 + (size_t)(nBase + col) * DIM;
        #pragma unroll
        for (int kc = 0; kc < 4; ++kc) {
            afrag[kc].h[0] = *(const v8h*)(arow + kc * 32 + hi * 8);
            afrag[kc].h[1] = *(const v8h*)(arow + kc * 32 + 16 + hi * 8);
        }
    }

    float dacc[NVIEWS][8];
    #pragma unroll
    for (int v = 0; v < NVIEWS; ++v)
        #pragma unroll
        for (int j = 0; j < 8; ++j) dacc[v][j] = 0.0f;

    // C layout: VGPR j <-> row nBase + hi*8 + j, column mBase + col.
    const size_t sbase = (size_t)(nBase + hi * 8) * NROWS + col;

    // Prologue: fill buffer 0 with the first tile.
    copy_tile(smem[0], Hv16, mStart, tid);

    for (int t = 0; t < nTiles; ++t) {
        const int cur   = t & 1;
        const int mBase = mStart + t * 16;

        // Writes into smem[cur] (issued last iteration / prologue) complete:
        wait_async_copies();
        __syncthreads();   // ...and visible to all waves; reads of smem[cur^1]
                           // from iteration t-1 are done -> safe to refill it.
        if (t + 1 < nTiles)
            copy_tile(smem[cur ^ 1], Hv16, mBase + 16, tid);

        // S tile (streamed once, non-temporal; 64B-coalesced per half-wave).
        float w[8];
        #pragma unroll
        for (int j = 0; j < 8; ++j)
            w[j] = 1.0f - __builtin_nontemporal_load(
                              S + sbase + (size_t)j * NROWS + mBase);

        #pragma unroll
        for (int v = 0; v < NVIEWS; ++v) {
            // B fragments (32x16 f16) from LDS: lane holds column col; per
            // K-chunk the lane's halves are [kc*32 + hi*8, +8) and [kc*32+16 + hi*8, +8).
            const _Float16* brow = &smem[cur][(v * 16 + col) * PADW + hi * 8];
            v8f c = {};
            #pragma unroll
            for (int kc = 0; kc < 4; ++kc) {
                V16 b;
                b.h[0] = *(const v8h*)(brow + kc * 32);
                b.h[1] = *(const v8h*)(brow + kc * 32 + 16);
                c = __builtin_amdgcn_wmma_f32_16x16x32_f16(
                        /*neg_a=*/false, afrag[kc].v,
                        /*neg_b=*/false, b.v,
                        /*c_mod=*/(short)0, c,
                        /*reuse_a=*/false, /*reuse_b=*/false);
            }
            // Epilogue: (1-S) * exp(sim/T); exp is TRANS -> co-executes.
            #pragma unroll
            for (int j = 0; j < 8; ++j)
                dacc[v][j] += w[j] * __expf(c[j] * INV_T);
        }
    }

    // Row totals: sum the 16 columns held across each half-wave.
    #pragma unroll
    for (int v = 0; v < NVIEWS; ++v) {
        #pragma unroll
        for (int j = 0; j < 8; ++j) {
            float tsum = half16_sum(dacc[v][j]);
            if (col == 0)  // lane 0 -> rows nBase+j ; lane 16 -> rows nBase+8+j
                denomP[((size_t)blockIdx.y * NVIEWS + v) * NROWS
                       + nBase + hi * 8 + j] = tsum;
        }
    }
}

// --- Kernel 4: deterministic final reduction -> scalar ---------------------
__global__ void sgcl_finalize(const float* __restrict__ denomP,
                              const float* __restrict__ pos,
                              float* __restrict__ out) {
    __shared__ float red[8];
    float acc = 0.0f;
    for (int i = threadIdx.x; i < NVIEWS * NROWS; i += 256) {
        float d = 0.0f;
        #pragma unroll
        for (int s = 0; s < MSPLIT; ++s)
            d += denomP[(size_t)s * NVIEWS * NROWS + i];
        acc += logf(fmaxf(d, EPS_D)) - pos[i] * INV_T;
    }
    acc = wave32_sum(acc);
    const int wave = threadIdx.x >> 5;
    const int lane = threadIdx.x & 31;
    if (lane == 0) red[wave] = acc;
    __syncthreads();
    if (threadIdx.x == 0) {
        float t = 0.0f;
        #pragma unroll
        for (int w = 0; w < 8; ++w) t += red[w];
        out[0] = t / (float)(NROWS * NVIEWS);
    }
}

// ---------------------------------------------------------------------------
extern "C" void kernel_launch(void* const* d_in, const int* in_sizes, int n_in,
                              void* d_out, int out_size, void* d_ws, size_t ws_size,
                              hipStream_t stream) {
    const float* Hc = (const float*)d_in[0];   // [8192,128]
    const float* S  = (const float*)d_in[1];   // [8192,8192]
    const float* Hv = (const float*)d_in[2];   // [3,8192,128]

    char* ws = (char*)d_ws;
    _Float16* Hc16 = (_Float16*)ws;                                     // 2 MB
    _Float16* Hv16 = (_Float16*)(ws + (size_t)NROWS * DIM * 2);         // 6 MB
    float* denomP  = (float*)(ws + (size_t)(NVIEWS + 1) * NROWS * DIM * 2);
    float* pos     = denomP + (size_t)MSPLIT * NVIEWS * NROWS;

    sgcl_normalize_f16<<<(4 * NROWS) / 8, 256, 0, stream>>>(Hc, Hv, Hc16, Hv16);
    sgcl_positives<<<(NVIEWS * NROWS) / 8, 256, 0, stream>>>(Hc, Hv, pos);
    sgcl_denominator<<<dim3(NROWS / 128, MSPLIT), 256, 0, stream>>>(
        S, Hc16, Hv16, denomP);
    sgcl_finalize<<<1, 256, 0, stream>>>(denomP, pos, (float*)d_out);
}
